// MODJSCC_MoE_Faithful_2_73340861546868
// MI455X (gfx1250) — matrix operand order repaired and checked by reading, not compiled, not run on hardware
//
#include <hip/hip_runtime.h>
#include <hip/hip_bf16.h>
#include <math.h>

// ---------------------------------------------------------------------------
// MoE-JSCC forward pass for MI455X (gfx1250, wave32, WMMA bf16 16x16x32)
// Weights pre-transposed to bf16 [N,K]; A loaded as float4 pairs + pack-cvt.
// Each wave computes a 16x64 C strip (4 accumulators) to amortize A loads.
// ---------------------------------------------------------------------------

typedef __attribute__((ext_vector_type(16))) __bf16 v16bf;
typedef __attribute__((ext_vector_type(8)))  float  v8f;

#define BB  8192
#define DD  256
#define TT  6
#define NSY 64
#define PKW (NSY * 6)   // packed bit-logit row width = 384

// ----------------------------- RNG helpers ---------------------------------
__device__ __forceinline__ uint32_t mix3(uint32_t a, uint32_t b, uint32_t c) {
  uint32_t h = a * 0x9E3779B1u;
  h ^= b + 0x85EBCA77u + (h << 6) + (h >> 2);
  h *= 0xC2B2AE3Du;
  h ^= c + 0x27D4EB2Fu + (h << 6) + (h >> 2);
  h ^= h >> 16; h *= 0x7FEB352Du; h ^= h >> 15; h *= 0x846CA68Bu; h ^= h >> 16;
  return h;
}
__device__ __forceinline__ float u01(uint32_t h) {
  return ((h >> 8) + 0.5f) * (1.0f / 16777216.0f);   // (0,1)
}
__device__ __forceinline__ float gumbel_from(uint32_t h) {
  float u = u01(h);
  return -__logf(-__logf(u));
}
__device__ __forceinline__ float softplus_f(float x) {
  return (x > 20.0f) ? x : log1pf(__expf(x));
}
__device__ __forceinline__ float4 addf4(float4 a, float4 b) {
  return make_float4(a.x + b.x, a.y + b.y, a.z + b.z, a.w + b.w);
}

// ----------------------- weight prep: fp32[K,N] -> bf16[N,K] ----------------
__global__ void wt_prep_kernel(const float* __restrict__ W,
                               __bf16* __restrict__ Wt, int K, int N) {
  int i = blockIdx.x * blockDim.x + threadIdx.x;
  if (i >= K * N) return;
  int k = i / N, n = i - k * N;
  Wt[(size_t)n * K + k] = (__bf16)W[i];
}

// ----------------------------- WMMA GEMM -----------------------------------
// C[M,N] = epi( A_eff[M,KD] @ Wt^T + bias )   Wt is bf16 [N, KD].
// Requires: M % 16 == 0, KD % 32 == 0. One wave -> 16x64 strip of C.
//
// AMODE: 0 plain A[M,KD]
//        1 A_eff[m,k] = y[m/T, k] + frame_emb[m%T, k]     (KD == 256)
//        3 A_eff[m,k] = k<256 ? y[m,k] : fused[m,k-256]   (KD == 512)
// EPI:   0 store                        1 relu store
//        2 store + (uniform(m,n)-0.5)   (quantization noise)
//        3 atomicAdd(fused[m/T,n], mask[m,e]*v/T)         (decode mix)
//        4 packed[m,(n/bps)*6+n%bps] += mask[m,e]*v       (encode mix)
//        5 relu(v + log(1/nvar[m])*A2[n])                 (concat-logsnr fold)
template<int AMODE, int EPI, int KD>
__global__ void gemm_wmma_kernel(const float* __restrict__ A,
                                 const float* __restrict__ A2,
                                 const float* __restrict__ nvar,
                                 const __bf16* __restrict__ Wt,
                                 const float* __restrict__ bias,
                                 float* __restrict__ C,
                                 const float* __restrict__ mask,
                                 int M, int N, int e, uint32_t seed) {
  const int lane = threadIdx.x & 31;
  const int wave = threadIdx.x >> 5;
  const int tiles_n = (N + 63) >> 6;              // groups of 64 columns
  const int tiles_m = M >> 4;
  const int tile = blockIdx.x * 8 + wave;
  if (tile >= tiles_m * tiles_n) return;          // wave-uniform exit
  const int m0 = (tile / tiles_n) << 4;
  const int ng = (tile % tiles_n) << 6;           // base column of 64-strip

  const int hf  = lane >> 4;                      // lane half (0/1)
  const int l16 = lane & 15;
  const int am  = m0 + l16;                       // A row owned by this lane

  int bn[4];
  const __bf16* wrow[4];
#pragma unroll
  for (int c = 0; c < 4; ++c) {
    int n = ng + (c << 4) + l16;                  // B col owned by this lane
    bn[c] = n;
    int nc = (n < N) ? n : (N - 1);               // clamp (loads in-bounds)
    wrow[c] = Wt + (size_t)nc * KD;
  }

  const float* ar0;
  const float* ar1 = nullptr;
  if (AMODE == 1) {
    int b = am / TT, t = am - b * TT;
    ar0 = A + (size_t)b * DD;
    ar1 = A2 + (size_t)t * DD;
  } else if (AMODE == 3) {
    ar0 = A + (size_t)am * DD;
    ar1 = A2 + (size_t)am * DD;
  } else {
    ar0 = A + (size_t)am * KD;
  }

  v8f acc0 = {}, acc1 = {}, acc2 = {}, acc3 = {};
#pragma unroll 2
  for (int k0 = 0; k0 < KD; k0 += 32) {
    const int ks0 = k0 + (hf << 4);               // first 8-float segment
    const int ks1 = ks0 + 8;                      // second 8-float segment
    float4 f0, f1, f2, f3;
    if (AMODE == 0) {
      f0 = *(const float4*)(ar0 + ks0);
      f1 = *(const float4*)(ar0 + ks0 + 4);
      f2 = *(const float4*)(ar0 + ks1);
      f3 = *(const float4*)(ar0 + ks1 + 4);
    } else if (AMODE == 1) {
      f0 = addf4(*(const float4*)(ar0 + ks0),     *(const float4*)(ar1 + ks0));
      f1 = addf4(*(const float4*)(ar0 + ks0 + 4), *(const float4*)(ar1 + ks0 + 4));
      f2 = addf4(*(const float4*)(ar0 + ks1),     *(const float4*)(ar1 + ks1));
      f3 = addf4(*(const float4*)(ar0 + ks1 + 4), *(const float4*)(ar1 + ks1 + 4));
    } else {  // AMODE == 3: segments never straddle the 256 boundary
      const float* p0 = (ks0 < DD) ? (ar0 + ks0) : (ar1 + (ks0 - DD));
      const float* p1 = (ks1 < DD) ? (ar0 + ks1) : (ar1 + (ks1 - DD));
      f0 = ((const float4*)p0)[0]; f1 = ((const float4*)p0)[1];
      f2 = ((const float4*)p1)[0]; f3 = ((const float4*)p1)[1];
    }
    v16bf af;
    af[0]  = (__bf16)f0.x; af[1]  = (__bf16)f0.y; af[2]  = (__bf16)f0.z; af[3]  = (__bf16)f0.w;
    af[4]  = (__bf16)f1.x; af[5]  = (__bf16)f1.y; af[6]  = (__bf16)f1.z; af[7]  = (__bf16)f1.w;
    af[8]  = (__bf16)f2.x; af[9]  = (__bf16)f2.y; af[10] = (__bf16)f2.z; af[11] = (__bf16)f2.w;
    af[12] = (__bf16)f3.x; af[13] = (__bf16)f3.y; af[14] = (__bf16)f3.z; af[15] = (__bf16)f3.w;

    // B fragments: k = k0 + 16*half + j -> 16 contiguous bf16 (32B aligned)
    const int kb = k0 + (hf << 4);
    v16bf b0 = *(const v16bf*)(wrow[0] + kb);
    v16bf b1 = *(const v16bf*)(wrow[1] + kb);
    v16bf b2 = *(const v16bf*)(wrow[2] + kb);
    v16bf b3 = *(const v16bf*)(wrow[3] + kb);

    acc0 = __builtin_amdgcn_wmma_f32_16x16x32_bf16(false, af, false, b0, (short)0, acc0, false, false);
    acc1 = __builtin_amdgcn_wmma_f32_16x16x32_bf16(false, af, false, b1, (short)0, acc1, false, false);
    acc2 = __builtin_amdgcn_wmma_f32_16x16x32_bf16(false, af, false, b2, (short)0, acc2, false, false);
    acc3 = __builtin_amdgcn_wmma_f32_16x16x32_bf16(false, af, false, b3, (short)0, acc3, false, false);
  }

  v8f accs[4] = {acc0, acc1, acc2, acc3};
#pragma unroll
  for (int c = 0; c < 4; ++c) {
    int n = bn[c];
    if (n >= N) continue;
#pragma unroll
    for (int r = 0; r < 8; ++r) {
      int m = m0 + r + (hf << 3);
      float v = accs[c][r] + bias[n];
      if (EPI == 0) {
        C[(size_t)m * N + n] = v;
      } else if (EPI == 1) {
        C[(size_t)m * N + n] = fmaxf(v, 0.0f);
      } else if (EPI == 2) {
        v += u01(mix3(seed, (uint32_t)m, (uint32_t)n)) - 0.5f;
        C[(size_t)m * N + n] = v;
      } else if (EPI == 3) {
        int b = m / TT;
        float w = mask[(size_t)m * 3 + e] * (1.0f / (float)TT);
        atomicAdd(&C[(size_t)b * DD + n], w * v);
      } else if (EPI == 4) {
        int bps = 2 * (e + 1);
        int s = n / bps, j = n - s * bps;
        float w = mask[(size_t)m * 3 + e];
        C[(size_t)m * PKW + s * 6 + j] += w * v;
      } else {  // EPI == 5: rank-1 logsnr column + relu
        v += __logf(1.0f / nvar[m]) * A2[n];
        C[(size_t)m * N + n] = fmaxf(v, 0.0f);
      }
    }
  }
}

// ----------------------------- small kernels -------------------------------
__global__ void zero_kernel(float* __restrict__ p, int n) {
  int i = blockIdx.x * blockDim.x + threadIdx.x;
  if (i < n) p[i] = 0.0f;
}

// One block (256 threads) per batch row: router features + hard gumbel route.
__global__ void router_kernel(const float* __restrict__ zt,
                              const float* __restrict__ rawsig,
                              const float* __restrict__ nvar,
                              const float* __restrict__ rs_w,   // [4,3]
                              const float* __restrict__ rs_b,   // [3]
                              float* __restrict__ mask,         // [B*T,3]
                              uint32_t seed) {
  __shared__ float s1[256], s2[256], s3[256];
  __shared__ float feat[4];
  int b = blockIdx.x, k = threadIdx.x;
  float z  = zt[(size_t)b * DD + k];
  float sp = softplus_f(rawsig[(size_t)b * DD + k]);
  float ls = log1pf(fabsf(sp) + 1e-6f);
  s1[k] = fabsf(z); s2[k] = ls; s3[k] = ls * ls;
  __syncthreads();
  for (int off = 128; off > 0; off >>= 1) {
    if (k < off) { s1[k] += s1[k + off]; s2[k] += s2[k + off]; s3[k] += s3[k + off]; }
    __syncthreads();
  }
  if (k == 0) {
    feat[0] = __logf(1.0f / nvar[b]);
    feat[1] = s1[0] * (1.0f / DD);
    float mls = s2[0] * (1.0f / DD);
    feat[2] = mls;
    feat[3] = s3[0] * (1.0f / DD) - mls * mls;
  }
  __syncthreads();
  if (k < TT) {
    int row = b * TT + k;
    float best = -1e30f; int bi = 0;
    for (int ex = 0; ex < 3; ++ex) {
      float lg = rs_b[ex];
      for (int i = 0; i < 4; ++i) lg += feat[i] * rs_w[i * 3 + ex];
      lg += gumbel_from(mix3(seed, (uint32_t)row, (uint32_t)ex));
      if (lg > best) { best = lg; bi = ex; }   // argmax == ST one-hot forward
    }
    for (int ex = 0; ex < 3; ++ex)
      mask[(size_t)row * 3 + ex] = (ex == bi) ? 1.0f : 0.0f;
  }
}

// Per (b,t,s): hard gumbel-sigmoid bits -> QAM constellation -> AWGN -> rxf.
__global__ void symbol_kernel(const float* __restrict__ packed,
                              const float* __restrict__ mask,
                              const float* __restrict__ nvar,
                              float* __restrict__ rxf,
                              uint32_t seed) {
  int idx = blockIdx.x * blockDim.x + threadIdx.x;
  if (idx >= BB * TT * NSY) return;
  int row = idx / NSY, s = idx - row * NSY;
  int b = row / TT;
  const float* mr = mask + (size_t)row * 3;
  int e = (mr[1] > 0.5f) ? 1 : ((mr[2] > 0.5f) ? 2 : 0);
  int bps = 2 * (e + 1);
  int bI = bps >> 1;                       // == bQ (bps even)
  float Ii = 0.0f, Qi = 0.0f;
  for (int j = 0; j < bps; ++j) {
    float l = packed[(size_t)row * PKW + s * 6 + j];
    float g = gumbel_from(mix3(seed, (uint32_t)idx, (uint32_t)j));
    float y = 1.0f / (1.0f + __expf(-(l + g)));   // tau = 1
    float bit = (y > 0.5f) ? 1.0f : 0.0f;         // ST hard forward
    if (j < bI) Ii = Ii * 2.0f + bit; else Qi = Qi * 2.0f + bit;
  }
  float L = (float)(1 << bI);
  float inv_norm = rsqrtf(2.0f * (L * L - 1.0f) / 3.0f);
  float sI = (2.0f * Ii + 1.0f - L) * inv_norm;
  float sQ = (2.0f * Qi + 1.0f - L) * inv_norm;
  // Box-Muller AWGN
  uint32_t h1 = mix3(seed ^ 0xA5A5A5A5u, (uint32_t)idx, 1u);
  uint32_t h2 = mix3(seed ^ 0xA5A5A5A5u, (uint32_t)idx, 2u);
  float r  = sqrtf(-2.0f * __logf(u01(h1)));
  float th = 6.28318530718f * u01(h2);
  float sn = sqrtf(nvar[b]);
  rxf[(size_t)row * (2 * NSY) + 2 * s + 0] = sI + sn * r * cosf(th);
  rxf[(size_t)row * (2 * NSY) + 2 * s + 1] = sQ + sn * r * sinf(th);
}

// erf-based discrete entropy rate + final packing: out[b] = {logit0, logit1, rate}
__global__ void rate_out_kernel(const float* __restrict__ zt,
                                const float* __restrict__ rawsig,
                                const float* __restrict__ lgts,
                                float* __restrict__ out) {
  __shared__ float sr[256];
  int b = blockIdx.x, k = threadIdx.x;
  float z   = zt[(size_t)b * DD + k];
  float sig = softplus_f(rawsig[(size_t)b * DD + k]) + 1e-6f;
  float inv = 1.0f / (sig * 1.41421356237f);
  float p = 0.5f * (erff((z + 0.5f) * inv) - erff((z - 0.5f) * inv));
  p = fmaxf(p, 1e-12f);
  sr[k] = -__log2f(p);
  __syncthreads();
  for (int off = 128; off > 0; off >>= 1) {
    if (k < off) sr[k] += sr[k + off];
    __syncthreads();
  }
  if (k == 0) {
    out[b * 3 + 0] = lgts[b * 2 + 0];
    out[b * 3 + 1] = lgts[b * 2 + 1];
    out[b * 3 + 2] = sr[0];
  }
}

// ----------------------------- launch --------------------------------------
static inline int gemm_blocks(int M, int N) {
  int tiles = (M / 16) * ((N + 63) / 64);
  return (tiles + 7) / 8;
}

extern "C" void kernel_launch(void* const* d_in, const int* in_sizes, int n_in,
                              void* d_out, int out_size, void* d_ws, size_t ws_size,
                              hipStream_t stream) {
  (void)in_sizes; (void)n_in; (void)out_size; (void)ws_size;
  const float* cls    = (const float*)d_in[0];
  const float* nvar   = (const float*)d_in[1];
  const float* proj_w = (const float*)d_in[2];
  const float* proj_b = (const float*)d_in[3];
  const float* he_w1  = (const float*)d_in[4];
  const float* he_b1  = (const float*)d_in[5];
  const float* he_w2  = (const float*)d_in[6];
  const float* he_b2  = (const float*)d_in[7];
  const float* hd_w1  = (const float*)d_in[8];
  const float* hd_b1  = (const float*)d_in[9];
  const float* hd_w2  = (const float*)d_in[10];
  const float* hd_b2  = (const float*)d_in[11];
  const float* rs_w   = (const float*)d_in[12];
  const float* rs_b   = (const float*)d_in[13];
  const float* frame  = (const float*)d_in[16];
  const float* enc_w[3] = {(const float*)d_in[17], (const float*)d_in[19], (const float*)d_in[21]};
  const float* enc_b[3] = {(const float*)d_in[18], (const float*)d_in[20], (const float*)d_in[22]};
  const float* dec_w[3] = {(const float*)d_in[23], (const float*)d_in[25], (const float*)d_in[27]};
  const float* dec_b[3] = {(const float*)d_in[24], (const float*)d_in[26], (const float*)d_in[28]};
  const float* d1_w   = (const float*)d_in[29];
  const float* d1_b   = (const float*)d_in[30];
  const float* d2_w   = (const float*)d_in[31];
  const float* d2_b   = (const float*)d_in[32];
  float* out = (float*)d_out;

  // -------- fp32 workspace layout --------
  float* ws     = (float*)d_ws;
  float* y      = ws;                              // B*256
  float* zt     = y      + (size_t)BB * DD;        // B*256
  float* rawsig = zt     + (size_t)BB * DD;        // B*256
  float* hbuf   = rawsig + (size_t)BB * DD;        // B*128 (reused)
  float* maskb  = hbuf   + (size_t)BB * 128;       // B*T*3
  float* packed = maskb  + (size_t)BB * TT * 3;    // B*T*384
  float* rxf    = packed + (size_t)BB * TT * PKW;  // B*T*128
  float* fused  = rxf    + (size_t)BB * TT * 128;  // B*256
  float* dhid   = fused  + (size_t)BB * DD;        // B*256
  float* lgts   = dhid   + (size_t)BB * DD;        // B*2
  float* fend   = lgts   + (size_t)BB * 2;

  // -------- bf16 transposed-weight arena --------
  __bf16* arena = (__bf16*)fend;
  const int THR = 256;
  struct WSpec { const float* W; int K; int N; };
  WSpec specs[13] = {
    {proj_w, 768, 256}, {he_w1, 256, 128}, {he_w2, 128, 256},
    {hd_w1, 256, 128},  {hd_w2, 128, 256},
    {enc_w[0], 256, 128}, {enc_w[1], 256, 256}, {enc_w[2], 256, 384},
    {dec_w[0], 128, 256}, {dec_w[1], 128, 256}, {dec_w[2], 128, 256},
    {d1_w, 512, 256}, {d2_w, 256, 2},
  };
  __bf16* wt[13];
  {
    size_t off = 0;
    for (int i = 0; i < 13; ++i) {
      wt[i] = arena + off;
      int n = specs[i].K * specs[i].N;
      wt_prep_kernel<<<(n + THR - 1) / THR, THR, 0, stream>>>(
          specs[i].W, wt[i], specs[i].K, specs[i].N);
      off += (size_t)n;
      off = (off + 15) & ~(size_t)15;   // keep 32B alignment
    }
  }

  // 1) y = cls @ proj_w + b
  gemm_wmma_kernel<0, 0, 768><<<gemm_blocks(BB, DD), THR, 0, stream>>>(
      cls, nullptr, nullptr, wt[0], proj_b, y, nullptr, BB, DD, 0, 0u);
  // 2) h = relu(y @ he_w1[0:256] + logsnr*he_w1[256] + b)   (K=257 folded)
  gemm_wmma_kernel<0, 5, 256><<<gemm_blocks(BB, 128), THR, 0, stream>>>(
      y, he_w1 + (size_t)256 * 128, nvar, wt[1], he_b1, hbuf, nullptr, BB, 128, 0, 0u);
  // 3) z_tilde = h @ he_w2 + b + U(-0.5, 0.5)
  gemm_wmma_kernel<0, 2, 128><<<gemm_blocks(BB, DD), THR, 0, stream>>>(
      hbuf, nullptr, nullptr, wt[2], he_b2, zt, nullptr, BB, DD, 0, 0x11111111u);
  // 4) h = relu(z_tilde @ hd_w1 + b)
  gemm_wmma_kernel<0, 1, 256><<<gemm_blocks(BB, 128), THR, 0, stream>>>(
      zt, nullptr, nullptr, wt[3], hd_b1, hbuf, nullptr, BB, 128, 0, 0u);
  // 5) raw_sigma = h @ hd_w2 + b
  gemm_wmma_kernel<0, 0, 128><<<gemm_blocks(BB, DD), THR, 0, stream>>>(
      hbuf, nullptr, nullptr, wt[4], hd_b2, rawsig, nullptr, BB, DD, 0, 0u);
  // 6) router: hard one-hot route mask
  router_kernel<<<BB, THR, 0, stream>>>(zt, rawsig, nvar, rs_w, rs_b, maskb, 0x22222222u);
  // 7) zero packed bit-logits and fused accumulator
  {
    int n1 = BB * TT * PKW;
    zero_kernel<<<(n1 + THR - 1) / THR, THR, 0, stream>>>(packed, n1);
    int n2 = BB * DD;
    zero_kernel<<<(n2 + THR - 1) / THR, THR, 0, stream>>>(fused, n2);
  }
  // 8) per-expert encoders (A = y[b] + frame_emb[t]), route-masked packed store
  {
    gemm_wmma_kernel<1, 4, 256><<<gemm_blocks(BB * TT, 128), THR, 0, stream>>>(
        y, frame, nullptr, wt[5], enc_b[0], packed, maskb, BB * TT, 128, 0, 0u);
    gemm_wmma_kernel<1, 4, 256><<<gemm_blocks(BB * TT, 256), THR, 0, stream>>>(
        y, frame, nullptr, wt[6], enc_b[1], packed, maskb, BB * TT, 256, 1, 0u);
    gemm_wmma_kernel<1, 4, 256><<<gemm_blocks(BB * TT, 384), THR, 0, stream>>>(
        y, frame, nullptr, wt[7], enc_b[2], packed, maskb, BB * TT, 384, 2, 0u);
  }
  // 9) bits -> constellation -> AWGN -> rxf
  {
    int n = BB * TT * NSY;
    symbol_kernel<<<(n + THR - 1) / THR, THR, 0, stream>>>(packed, maskb, nvar, rxf, 0x33333333u);
  }
  // 10) per-expert decoders, fused route-mix + temporal mean into `fused`
  for (int e = 0; e < 3; ++e) {
    gemm_wmma_kernel<0, 3, 128><<<gemm_blocks(BB * TT, DD), THR, 0, stream>>>(
        rxf, nullptr, nullptr, wt[8 + e], dec_b[e], fused, maskb, BB * TT, DD, e, 0u);
  }
  // 11) dhid = relu(concat(y, fused) @ d1_w + b)
  gemm_wmma_kernel<3, 1, 512><<<gemm_blocks(BB, DD), THR, 0, stream>>>(
      y, fused, nullptr, wt[11], d1_b, dhid, nullptr, BB, DD, 0, 0u);
  // 12) logits = dhid @ d2_w + b   (N=2, clamped B loads)
  gemm_wmma_kernel<0, 0, 256><<<gemm_blocks(BB, 2), THR, 0, stream>>>(
      dhid, nullptr, nullptr, wt[12], d2_b, lgts, nullptr, BB, 2, 0, 0u);
  // 13) rate + final packing
  rate_out_kernel<<<BB, THR, 0, stream>>>(zt, rawsig, lgts, out);
}
